// NodeModel_56427280335505
// MI455X (gfx1250) — matrix operand order repaired and checked by reading, not compiled, not run on hardware
//
#include <hip/hip_runtime.h>
#include <math.h>

typedef __attribute__((ext_vector_type(16))) _Float16 v16h;
typedef __attribute__((ext_vector_type(8)))  _Float16 v8h;
typedef __attribute__((ext_vector_type(8)))  float    v8f;

#define HDIM 128
#define NLOG2F 0.69314718055994530942f

__device__ __forceinline__ float ssp_f(float x) {
    // softplus(x) - log(2), numerically stable
    return fmaxf(x, 0.0f) + log1pf(__expf(-fabsf(x))) - NLOG2F;
}

__device__ __forceinline__ v16h cat8(v8h lo, v8h hi) {
    return __builtin_shufflevector(lo, hi, 0, 1, 2, 3, 4, 5, 6, 7,
                                           8, 9, 10, 11, 12, 13, 14, 15);
}

__global__ void zero_f4_kernel(float4* __restrict__ p, int n4) {
    int i = blockIdx.x * blockDim.x + threadIdx.x;
    if (i < n4) p[i] = make_float4(0.f, 0.f, 0.f, 0.f);
}

// Transpose + convert W [K,128] f32 -> Wt [128,K] f16 (coalesced writes; tiny, L2-resident)
__global__ void pack_w_kernel(const float* __restrict__ W, _Float16* __restrict__ Wt, int K) {
    int i = blockIdx.x * blockDim.x + threadIdx.x;   // i indexes output: n*K + k
    if (i >= K * HDIM) return;
    int n = i / K;
    int k = i - n * K;
    Wt[i] = (_Float16)W[(size_t)k * HDIM + n];
}

// One thread per (edge, 4-float group): float4 load + 4 global f32 atomics.
__global__ void scatter_add_kernel(const float* __restrict__ edge_attr,
                                   const int* __restrict__ col,
                                   float* __restrict__ agg,
                                   int E) {
    int i = blockIdx.x * blockDim.x + threadIdx.x;
    int total = E * (HDIM / 4);
    if (i >= total) return;
    int e = i >> 5;          // HDIM/4 == 32 groups per edge
    int g = i & 31;
    int c = col[e];
    float4 v = reinterpret_cast<const float4*>(edge_attr)[(size_t)e * (HDIM / 4) + g];
    float* dst = agg + (size_t)c * HDIM + (size_t)g * 4;
    atomicAdd(dst + 0, v.x);
    atomicAdd(dst + 1, v.y);
    atomicAdd(dst + 2, v.z);
    atomicAdd(dst + 3, v.w);
}

// Fused: out = ssp(layer_norm(A @ W + bias, gamma, beta))
// A is [N, KDIM]: f16 when A_HALF (contiguous b128 fragment loads), else f32 [A0|A1] concat
// with in-register cvt. Wt is [128][KDIM] f16 (transposed) -> B fragment is two runs of 8 contiguous
// halves = two b128 loads. 256 threads = 8 wave32s; block -> 16 rows x 128 cols; wave w ->
// 16x16 tile at column 16*w via V_WMMA_F32_16X16X32_F16 (f32 accumulate).
template<int KDIM, bool A_HALF, bool OUT_HALF>
__global__ __launch_bounds__(256)
void layer_kernel(const void* __restrict__ A0v,
                  const void* __restrict__ A1v,
                  const _Float16* __restrict__ Wt,
                  const float* __restrict__ bias,
                  const float* __restrict__ gamma,
                  const float* __restrict__ beta,
                  void* __restrict__ outv,
                  int N)
{
    __shared__ float tile[16][HDIM + 4];
    __shared__ float psum[16][16];
    __shared__ float psq[16][16];
    __shared__ float mu_s[16];
    __shared__ float rs_s[16];

    const int tid  = threadIdx.x;
    const int lane = tid & 31;
    const int wave = tid >> 5;
    const int half = lane >> 4;   // 0: lanes 0-15, 1: lanes 16-31
    const int l16  = lane & 15;
    const int m0   = blockIdx.x * 16;
    const int n0   = wave * 16;
    const int ncol = n0 + l16;    // output column 0..127

    int rowA = m0 + l16;
    if (rowA >= N) rowA = N - 1;            // WMMA needs EXEC all-ones: clamp, don't predicate
    const size_t aBase = (size_t)rowA * HDIM;       // element units (f32 or f16)
    const _Float16* Bcol = Wt + (size_t)ncol * KDIM; // this lane's W column, f16

    v8f acc = {0.f, 0.f, 0.f, 0.f, 0.f, 0.f, 0.f, 0.f};

    #pragma unroll
    for (int kb = 0; kb < KDIM; kb += 32) {
        v16h a, b;
        if (A_HALF) {
            // A 16x32 f16 layout: a[0..7]=K(kb+8h..), a[8..15]=K(kb+16+8h..) — contiguous
            const _Float16* Ar = (const _Float16*)A0v + aBase + kb + 8 * half;
            a = cat8(*(const v8h*)Ar, *(const v8h*)(Ar + 16));
        } else {
            const float* Arow;
            int koff;
            if (KDIM == 256 && kb >= HDIM) {  // implicit concat [x | agg]; 128%32==0 no straddle
                Arow = (const float*)A1v + aBase;
                koff = kb - HDIM;
            } else {
                Arow = (const float*)A0v + aBase;
                koff = kb;
            }
            #pragma unroll
            for (int j = 0; j < 8; ++j)
                a[j] = (_Float16)Arow[koff + 8 * half + j];
            #pragma unroll
            for (int j = 0; j < 8; ++j)
                a[8 + j] = (_Float16)Arow[koff + 16 + 8 * half + j];
        }
        {
            // B 32x16 layout: b[0..7]=K(kb+8h..), b[8..15]=K(kb+16+8h..) (matches A);
            // transposed-f16 W makes these two runs of 8 contiguous halves = two b128 loads.
            const _Float16* Br = Bcol + kb + 8 * half;
            b = cat8(*(const v8h*)Br, *(const v8h*)(Br + 16));
        }
        acc = __builtin_amdgcn_wmma_f32_16x16x32_f16(
            /*neg_a=*/false, a, /*neg_b=*/false, b,
            /*c_mod=*/(short)0, acc, /*reuse_a=*/false, /*reuse_b=*/false);
    }

    // bias add + spill the 16x128 block to LDS (C/D layout: VGPR v -> row v+8*half, col ncol)
    const float bv = bias[ncol];
    #pragma unroll
    for (int v = 0; v < 8; ++v)
        tile[v + 8 * half][ncol] = acc[v] + bv;
    __syncthreads();

    // LayerNorm partial reduction: thread t -> row t/16, 8 columns starting at (t%16)*8
    {
        const int row = tid >> 4;
        const int seg = tid & 15;
        float s = 0.f, q = 0.f;
        #pragma unroll
        for (int j = 0; j < 8; ++j) {
            float v = tile[row][seg * 8 + j];
            s += v;
            q += v * v;
        }
        psum[row][seg] = s;
        psq[row][seg]  = q;
    }
    __syncthreads();
    if (tid < 16) {
        float s = 0.f, q = 0.f;
        #pragma unroll
        for (int j = 0; j < 16; ++j) { s += psum[tid][j]; q += psq[tid][j]; }
        float mu  = s * (1.0f / HDIM);
        float var = q * (1.0f / HDIM) - mu * mu;   // biased var, matches jnp.var
        mu_s[tid] = mu;
        rs_s[tid] = rsqrtf(var + 1e-5f);
    }
    __syncthreads();
    {
        const int row = tid >> 4;
        const int seg = tid & 15;
        const int grow = m0 + row;
        if (grow < N) {
            const float mu = mu_s[row], rs = rs_s[row];
            const size_t obase = (size_t)grow * HDIM;
            #pragma unroll
            for (int j = 0; j < 8; ++j) {
                int c = seg * 8 + j;
                float y = (tile[row][c] - mu) * rs * gamma[c] + beta[c];
                float r = ssp_f(y);
                if (OUT_HALF)
                    ((_Float16*)outv)[obase + c] = (_Float16)r;
                else
                    ((float*)outv)[obase + c] = r;
            }
        }
    }
}

extern "C" void kernel_launch(void* const* d_in, const int* in_sizes, int n_in,
                              void* d_out, int out_size, void* d_ws, size_t ws_size,
                              hipStream_t stream) {
    const float* x    = (const float*)d_in[0];
    const int*   eidx = (const int*)d_in[1];
    const float* ea   = (const float*)d_in[2];
    const float* W1   = (const float*)d_in[3];
    const float* b1   = (const float*)d_in[4];
    const float* g1   = (const float*)d_in[5];
    const float* be1  = (const float*)d_in[6];
    const float* W2   = (const float*)d_in[7];
    const float* b2   = (const float*)d_in[8];
    const float* g2   = (const float*)d_in[9];
    const float* be2  = (const float*)d_in[10];
    const float* W3   = (const float*)d_in[11];
    const float* b3   = (const float*)d_in[12];
    const float* g3   = (const float*)d_in[13];
    const float* be3  = (const float*)d_in[14];

    const int N = in_sizes[0] / HDIM;       // 100000
    const int E = in_sizes[2] / HDIM;       // 600000
    const int* col = eidx + E;              // edge_index row 1 (destinations)

    const size_t NH = (size_t)N * HDIM;
    float*    agg = (float*)d_ws;                 // [N,128] f32 (atomic accumulation)
    _Float16* h1  = (_Float16*)(agg + NH);        // [N,128] f16
    _Float16* h2  = h1 + NH;                      // [N,128] f16
    _Float16* W1t = h2 + NH;                      // [128,256] f16 transposed
    _Float16* W2t = W1t + 256 * HDIM;             // [128,128] f16 transposed
    _Float16* W3t = W2t + 128 * HDIM;             // [128,128] f16 transposed
    float*    outp = (float*)d_out;

    // Pre-pack weights (tiny, ~128KB total; stays resident in L2)
    pack_w_kernel<<<(256 * HDIM + 255) / 256, 256, 0, stream>>>(W1, W1t, 256);
    pack_w_kernel<<<(128 * HDIM + 255) / 256, 256, 0, stream>>>(W2, W2t, 128);
    pack_w_kernel<<<(128 * HDIM + 255) / 256, 256, 0, stream>>>(W3, W3t, 128);

    {   // agg = 0
        int n4   = (int)(NH / 4);
        int grid = (n4 + 255) / 256;
        zero_f4_kernel<<<grid, 256, 0, stream>>>((float4*)agg, n4);
    }
    {   // agg += scatter(edge_attr, col)
        int total = E * (HDIM / 4);
        int grid  = (total + 255) / 256;
        scatter_add_kernel<<<grid, 256, 0, stream>>>(ea, col, agg, E);
    }
    const int gridN = (N + 15) / 16;
    layer_kernel<256, false, true ><<<gridN, 256, 0, stream>>>(x,  agg,     W1t, b1, g1, be1, h1,   N);
    layer_kernel<128, true,  true ><<<gridN, 256, 0, stream>>>(h1, nullptr, W2t, b2, g2, be2, h2,   N);
    layer_kernel<128, true,  false><<<gridN, 256, 0, stream>>>(h2, nullptr, W3t, b3, g3, be3, outp, N);
}
